// PFNet_27238682591749
// MI455X (gfx1250) — matrix operand order repaired and checked by reading, not compiled, not run on hardware
//
#include <hip/hip_runtime.h>

// ---------------------------------------------------------------------------
// PFNet forward for MI455X (gfx1250): bf16 WMMA GEMMs (v_wmma_f32_16x16x32_bf16)
// for dense layers, per-bin gram matrices and block-sparse SpMM. Fragments are
// fed with b128 LDS/global vector loads in contiguous k order (k = 16*hi + s).
// ---------------------------------------------------------------------------

typedef __bf16 bf16;
typedef __attribute__((ext_vector_type(16))) __bf16 v16bf;
typedef __attribute__((ext_vector_type(8)))  __bf16 v8bf;
typedef __attribute__((ext_vector_type(8)))  float  v8f;

#define NBINS 64
#define MAXPB 256
#define BATCH 4
#define NPTS  5000
#define PTOT  (BATCH * NPTS)

__device__ __forceinline__ v8f wmma_bf16(v16bf a, v16bf b, v8f c) {
  return __builtin_amdgcn_wmma_f32_16x16x32_bf16(false, a, false, b, (short)0, c,
                                                 false, false);
}

// merge the two contiguous 8-element runs of an A fragment (contiguous order:
// lane holds k = hi*16..hi*16+15)
__device__ __forceinline__ v16bf a_merge(v8bf lo, v8bf hilo) {
  return __builtin_shufflevector(lo, hilo, 0, 1, 2, 3, 4, 5, 6, 7, 8, 9, 10, 11,
                                 12, 13, 14, 15);
}

__device__ __forceinline__ float selu_f(float x) {
  const float sc = 1.0507009873554805f, al = 1.6732632423543772f;
  return x > 0.f ? sc * x : sc * al * (__expf(x) - 1.f);
}

// ---------------------------------------------------------------------------
// GEMM: out = act(A[M,Kpad](bf16) @ W + bias) * row_scale, W given transposed
// WT[Np,Kpad]. Block: 256 threads = 8 waves; tile 128(M) x 64(N); each wave
// computes 32x32 (4 WMMAs / K-step). K step 32 staged through LDS.
// act: 0=none, 1=SELU, 2=sigmoid
// ---------------------------------------------------------------------------
__global__ __launch_bounds__(256) void gemm_bias_act(
    const bf16* __restrict__ A, int lda, const bf16* __restrict__ WT, int ldwt,
    const float* __restrict__ bias, const float* __restrict__ row_scale,
    bf16* __restrict__ out_bf, float* __restrict__ out_f32, int ldo, int M,
    int Np, int Kpad, int act) {
  __shared__ bf16 As[128 * 40];  // [row][k], stride 40 to spread banks
  __shared__ bf16 Ws[64 * 48];   // [n][k] (transposed tile), stride 48
  const int t = threadIdx.x;
  const int lane = t & 31, wv = t >> 5;
  const int hi = lane >> 4, lr = lane & 15;
  const int m0 = blockIdx.x * 128, n0 = blockIdx.y * 64;
  const int wm = (wv & 3) * 32, wn = (wv >> 2) * 32;
  v8f acc[2][2];
  const v8f vzero = {0.f, 0.f, 0.f, 0.f, 0.f, 0.f, 0.f, 0.f};
#pragma unroll
  for (int i = 0; i < 2; ++i)
#pragma unroll
    for (int j = 0; j < 2; ++j) acc[i][j] = vzero;

  for (int k0 = 0; k0 < Kpad; k0 += 32) {
    {  // stage A tile 128x32 (2 threads / row, 32B each)
      int r = t >> 1, h = t & 1;
      int gm = m0 + r;
      uint4 v0 = {0u, 0u, 0u, 0u}, v1 = {0u, 0u, 0u, 0u};
      if (gm < M) {
        const bf16* src = A + (size_t)gm * lda + k0 + h * 16;
        v0 = *(const uint4*)(src);
        v1 = *(const uint4*)(src + 8);
        if (k0 + 32 < Kpad) __builtin_prefetch(src + 32, 0, 0);
      }
      *(uint4*)(As + r * 40 + h * 16) = v0;
      *(uint4*)(As + r * 40 + h * 16 + 8) = v1;
    }
    {  // stage W tile 64(n) x 32(k) from transposed weights
      int r = t >> 2, seg = t & 3;
      int gn = n0 + r;
      uint4 v = {0u, 0u, 0u, 0u};
      if (gn < Np) v = *(const uint4*)(WT + (size_t)gn * ldwt + k0 + seg * 8);
      *(uint4*)(Ws + r * 48 + seg * 8) = v;
    }
    __syncthreads();
    v16bf afrag[2], bfrag[2];
#pragma unroll
    for (int s = 0; s < 2; ++s) {
      const bf16* ap = As + (wm + s * 16 + lr) * 40 + hi * 16;
      afrag[s] = a_merge(*(const v8bf*)(ap), *(const v8bf*)(ap + 8));
      bfrag[s] = *(const v16bf*)(Ws + (wn + s * 16 + lr) * 48 + hi * 16);
    }
#pragma unroll
    for (int sm = 0; sm < 2; ++sm)
#pragma unroll
      for (int sn = 0; sn < 2; ++sn)
        acc[sm][sn] = wmma_bf16(afrag[sm], bfrag[sn], acc[sm][sn]);
    __syncthreads();
  }
#pragma unroll
  for (int sm = 0; sm < 2; ++sm)
#pragma unroll
    for (int sn = 0; sn < 2; ++sn) {
      int n = n0 + wn + sn * 16 + lr;
      if (n < Np) {
        float bval = bias ? bias[n] : 0.f;
#pragma unroll
        for (int v = 0; v < 8; ++v) {
          int m = m0 + wm + sm * 16 + v + hi * 8;
          if (m < M) {
            float val = acc[sm][sn][v] + bval;
            if (act == 1) val = selu_f(val);
            else if (act == 2) val = 1.f / (1.f + __expf(-val));
            if (row_scale) val *= row_scale[m];
            if (out_bf) out_bf[(size_t)m * ldo + n] = (bf16)val;
            if (out_f32) out_f32[(size_t)m * ldo + n] = val;
          }
        }
      }
    }
}

// ---------------------------------------------------------------------------
// Per-bin gram: dm[bb][m][n] = dot(feat[order[m]][64:256], feat[order[n]][64:256])
// grid.x = B*NBINS, grid.y = 16 (row tiles); 8 waves cover 16 col tiles.
// Fragments loaded directly from global with b128 vector loads.
// ---------------------------------------------------------------------------
__global__ __launch_bounds__(256) void gram_kernel(
    const bf16* __restrict__ feat,   // [PTOT,256] bf16 (cols 64.. are coords)
    const int* __restrict__ order,   // [B*NBINS*256] global rows
    float* __restrict__ dm) {        // [B*NBINS,256,256]
  const int bb = blockIdx.x, mt = blockIdx.y;
  const int t = threadIdx.x, lane = t & 31, wv = t >> 5;
  const int hi = lane >> 4, lr = lane & 15;
  const int* ord = order + bb * MAXPB;
  const int grow = ord[mt * 16 + lr];
  const bf16* arow = feat + (size_t)grow * 256 + 64;
#pragma unroll
  for (int cti = 0; cti < 2; ++cti) {
    const int ct = wv + cti * 8;
    const int gcol = ord[ct * 16 + lr];
    const bf16* brow = feat + (size_t)gcol * 256 + 64;
    v8f acc = {0.f, 0.f, 0.f, 0.f, 0.f, 0.f, 0.f, 0.f};
    for (int k0 = 0; k0 < 192; k0 += 32) {
      const bf16* ap = arow + k0 + hi * 16;
      v16bf av = a_merge(*(const v8bf*)(ap), *(const v8bf*)(ap + 8));
      v16bf bv = *(const v16bf*)(brow + k0 + hi * 16);
      acc = wmma_bf16(av, bv, acc);
    }
    float* out = dm + (size_t)bb * MAXPB * MAXPB;
#pragma unroll
    for (int v = 0; v < 8; ++v) {
      int m = mt * 16 + v + hi * 8;
      out[(size_t)m * MAXPB + ct * 16 + lr] = acc[v];
    }
  }
}

// ---------------------------------------------------------------------------
// Block-sparse SpMM: Out[order[m]][h] += sum_k A[bb][m][k] * Y[order[k]][h]
// grid.x = B*NBINS, grid.y = 16 (m tiles). Gathered Y slab staged *transposed*
// in LDS (ysT[h][k]) so B fragments are aligned v16bf LDS loads.
// ---------------------------------------------------------------------------
__global__ __launch_bounds__(256) void spmm_kernel(
    const bf16* __restrict__ Ab,     // [B*NBINS,256,256] bf16 adjacency
    const int* __restrict__ order,   // [B*NBINS*256]
    const bf16* __restrict__ Y,      // [PTOT, Hpad]
    float* __restrict__ Out,         // [PTOT, Hpad] (pre-zeroed)
    int Hpad) {
  __shared__ bf16 ysT[544 * 48];  // [h][k], k-stride 48
  __shared__ int ords[32];
  const int bb = blockIdx.x, mt = blockIdx.y;
  const int t = threadIdx.x, lane = t & 31, wv = t >> 5;
  const int hi = lane >> 4, lr = lane & 15;
  const int* ord = order + bb * MAXPB;
  const bf16* Abin = Ab + (size_t)bb * MAXPB * MAXPB;
  const int NT = Hpad / 16;
  const v8f vzero = {0.f, 0.f, 0.f, 0.f, 0.f, 0.f, 0.f, 0.f};
  v8f acc[5];
#pragma unroll
  for (int i = 0; i < 5; ++i) acc[i] = vzero;
  const int segs = Hpad / 8;
  for (int kc = 0; kc < 8; ++kc) {
    __syncthreads();
    if (t < 32) ords[t] = ord[kc * 32 + t];
    __syncthreads();
    for (int idx = t; idx < 32 * segs; idx += 256) {
      int r = idx / segs, sg = idx - r * segs;
      v8bf v = *(const v8bf*)(Y + (size_t)ords[r] * Hpad + sg * 8);
      int hb = sg * 8;
#pragma unroll
      for (int j = 0; j < 8; ++j) ysT[(hb + j) * 48 + r] = v[j];
    }
    __syncthreads();
    const bf16* ap = Abin + (size_t)(mt * 16 + lr) * MAXPB + kc * 32 + hi * 16;
    v16bf av = a_merge(*(const v8bf*)(ap), *(const v8bf*)(ap + 8));
#pragma unroll
    for (int ci = 0; ci < 5; ++ci) {
      int ct = wv + ci * 8;
      if (ct < NT) {  // wave-uniform: EXEC stays full for WMMA
        v16bf bv = *(const v16bf*)(ysT + (ct * 16 + lr) * 48 + hi * 16);
        acc[ci] = wmma_bf16(av, bv, acc[ci]);
      }
    }
  }
#pragma unroll
  for (int v = 0; v < 8; ++v) {
    int mslot = mt * 16 + v + hi * 8;
    int drow = ord[mslot];  // padding slots have zero A rows -> add 0
#pragma unroll
    for (int ci = 0; ci < 5; ++ci) {
      int ct = wv + ci * 8;
      if (ct < NT) atomicAdd(Out + (size_t)drow * Hpad + ct * 16 + lr, acc[ci][v]);
    }
  }
}

// --------------------------- small / glue kernels ---------------------------

__global__ void encode_kernel(const float* __restrict__ X, bf16* __restrict__ enc,
                              int P) {
  int idx = blockIdx.x * blockDim.x + threadIdx.x;
  int p = idx >> 5, c = idx & 31;
  if (p >= P) return;
  const float* xr = X + (size_t)p * 15;
  float v = 0.f;
  if (c < 12) {
    int id = (int)xr[0];
    v = (id == c) ? 1.f : 0.f;
  } else if (c < 26) {
    v = xr[1 + (c - 12)];
  }
  enc[(size_t)p * 32 + c] = (bf16)v;
}

// pack f32 [K,N] row-major -> bf16 transposed [Np,Kp] zero-padded
__global__ void pack_weight_t(const float* __restrict__ w, bf16* __restrict__ out,
                              int K, int N, int Kp, int Np) {
  int idx = blockIdx.x * blockDim.x + threadIdx.x;
  if (idx >= Kp * Np) return;
  int n = idx / Kp, k = idx - n * Kp;
  float v = (k < K && n < N) ? w[(size_t)k * N + n] : 0.f;
  out[idx] = (bf16)v;
}

__global__ void pack_bias(const float* __restrict__ b, float* __restrict__ out,
                          int N, int Np) {
  int n = blockIdx.x * blockDim.x + threadIdx.x;
  if (n < Np) out[n] = (n < N) ? b[n] : 0.f;
}

__global__ void zero_f32(float* __restrict__ p, int n) {
  int i = blockIdx.x * blockDim.x + threadIdx.x;
  if (i < n) p[i] = 0.f;
}

__global__ void bin_argmax(const float* __restrict__ xd, int* __restrict__ bin_idx,
                           int P) {
  int p = blockIdx.x * blockDim.x + threadIdx.x;
  if (p >= P) return;
  const float* r = xd + (size_t)p * 256;
  float best = r[0];
  int bi = 0;
  for (int j = 1; j < NBINS; ++j) {
    float v = r[j];
    if (v > best) { best = v; bi = j; }
  }
  bin_idx[p] = bi;
}

// one wave32 per (batch, bin): ordered compaction via ballot + prefix popcount
__global__ void build_order(const int* __restrict__ bin_idx, int* __restrict__ order,
                            int* __restrict__ counts) {
  const int bb = blockIdx.x;
  const int b = bb >> 6, bin = bb & 63;
  const int lane = threadIdx.x;
  int* ord = order + bb * MAXPB;
  int cnt = 0;
  for (int base = 0; base < NPTS; base += 32) {
    int n = base + lane;
    bool mem = (n < NPTS) && (bin_idx[b * NPTS + n] == bin);
    unsigned mask = (unsigned)__ballot(mem);
    int pre = __popc(mask & ((1u << lane) - 1u));
    if (mem) {
      int idx = cnt + pre;
      if (idx < MAXPB) ord[idx] = b * NPTS + n;
    }
    cnt += __popc(mask);
    if (cnt >= MAXPB) break;
  }
  int c = cnt < MAXPB ? cnt : MAXPB;
  if (lane == 0) counts[bb] = c;
  for (int j = c + lane; j < MAXPB; j += 32) ord[j] = b * NPTS;  // pad: adds zeros
}

// row-wise masked softmax -> cutoff -> exp(-soft); degree scatter-add
__global__ void adj_softmax(const float* __restrict__ dm, bf16* __restrict__ Ab,
                            const int* __restrict__ order,
                            const int* __restrict__ counts,
                            float* __restrict__ deg) {
  const int bb = blockIdx.x;
  const int r = threadIdx.x;
  const int cnt = counts[bb];
  const float* row = dm + ((size_t)bb * MAXPB + r) * MAXPB;
  bf16* arow = Ab + ((size_t)bb * MAXPB + r) * MAXPB;
  const bool vr = r < cnt;
  float mx = -1e30f;
  for (int j = 0; j < cnt; ++j) mx = fmaxf(mx, row[j]);
  float s = 0.f;
  for (int j = 0; j < cnt; ++j) s += __expf(row[j] - mx);
  float inv = (cnt > 0) ? 1.f / s : 0.f;
  float rsum = 0.f;
  for (int j = 0; j < MAXPB; ++j) {
    float a = 0.f;
    if (vr && j < cnt) {
      float soft = __expf(row[j] - mx) * inv;
      if (soft > 0.2f) { a = __expf(-soft); rsum += a; }
    }
    arow[j] = (bf16)a;
  }
  if (rsum != 0.f) atomicAdd(deg + order[bb * MAXPB + r], rsum);
}

__global__ void norm_kernel(const float* __restrict__ deg, float* __restrict__ nrm,
                            int P) {
  int p = blockIdx.x * blockDim.x + threadIdx.x;
  if (p < P) nrm[p] = rsqrtf(deg[p] + 1e-6f);
}

// gate*spmm*norm + (1-gate)*fhet -> bf16 (pad cols zeroed)
__global__ void ghconv_combine(const float* __restrict__ gate,
                               const float* __restrict__ fhet,
                               const float* __restrict__ spmm,
                               const float* __restrict__ nrm,
                               bf16* __restrict__ out, int P, int H, int Hpad) {
  int idx = blockIdx.x * blockDim.x + threadIdx.x;
  if (idx >= P * Hpad) return;
  int p = idx / Hpad, c = idx - p * Hpad;
  float v = 0.f;
  if (c < H) {
    float g = gate[(size_t)p * Hpad + c];
    float fh = fhet[(size_t)p * Hpad + c];
    float fo = spmm[(size_t)p * Hpad + c] * nrm[p];
    v = g * fo + (1.f - g) * fh;
  }
  out[idx] = (bf16)v;
}

__global__ void concat_kernel(const bf16* __restrict__ xm,
                              const float* __restrict__ outid,
                              bf16* __restrict__ xc, int P) {
  int idx = blockIdx.x * blockDim.x + threadIdx.x;
  if (idx >= P * 544) return;
  int p = idx / 544, c = idx - p * 544;
  bf16 v = (bf16)0.f;
  if (c < 512) v = xm[(size_t)p * 512 + c];
  else if (c < 520) v = (bf16)outid[(size_t)p * 8 + (c - 512)];
  xc[idx] = v;
}

__global__ void small_dense(const bf16* __restrict__ A, int lda,
                            const float* __restrict__ w, const float* __restrict__ b,
                            float* __restrict__ out, int P, int K, int NOUT) {
  int idx = blockIdx.x * blockDim.x + threadIdx.x;
  if (idx >= P * NOUT) return;
  int p = idx / NOUT, j = idx - p * NOUT;
  const bf16* ar = A + (size_t)p * lda;
  float s = b[j];
  for (int k = 0; k < K; ++k) s += (float)ar[k] * w[(size_t)k * NOUT + j];
  out[idx] = s;
}

__global__ void final_kernel(const float* __restrict__ X,
                             const float* __restrict__ outid,
                             const float* __restrict__ outmom,
                             const float* __restrict__ outch,
                             float* __restrict__ out, int P) {
  int idx = blockIdx.x * blockDim.x + threadIdx.x;
  if (idx >= P * 12) return;
  int p = idx / 12, c = idx - p * 12;
  float m = (X[(size_t)p * 15] != 0.f) ? 1.f : 0.f;
  float v;
  if (c < 8) v = outid[(size_t)p * 8 + c];
  else if (c < 11) v = X[(size_t)p * 15 + 2 + (c - 8)] + outmom[(size_t)p * 3 + (c - 8)];
  else v = outch[p];
  out[idx] = v * m;
}

// ---------------------------------------------------------------------------
// Host orchestration
// ---------------------------------------------------------------------------
extern "C" void kernel_launch(void* const* d_in, const int* in_sizes, int n_in,
                              void* d_out, int out_size, void* d_ws, size_t ws_size,
                              hipStream_t stream) {
  (void)in_sizes; (void)n_in; (void)out_size; (void)ws_size;
  const int P = PTOT;
  const float* X = (const float*)d_in[0];

  // flattened param order (setup_inputs dict insertion order)
  enum {
    I_DIST_W = 1, I_DIST_B, I1_W, I1_B, I2_W, I2_B, I3_W, I3_B,
    I1M_W, I1M_B, I2M_W, I2M_B, I3M_W, I3M_B,
    C1_WT, C1_BT, C1_WH, C1_TH, C2_WT, C2_BT, C2_WH, C2_TH,
    ID1_W, ID1_B, ID2_W, ID2_B, ID3_W, ID3_B,
    OID_W, OID_B, OCH_W, OCH_B,
    M1_W, M1_B, M2_W, M2_B, M3_W, M3_B, OM_W, OM_B
  };
  auto fin = [&](int i) { return (const float*)d_in[i]; };

  // workspace bump allocator
  char* wsb = (char*)d_ws;
  size_t off = 0;
  auto alloc = [&](size_t bytes) -> void* {
    off = (off + 255) & ~(size_t)255;
    void* p = wsb + off;
    off += bytes;
    return p;
  };

  // packed bf16 weights (transposed [Np,Kp])
  bf16* wd  = (bf16*)alloc(32 * 256 * 2);
  bf16* w1  = (bf16*)alloc(32 * 256 * 2);
  bf16* w2  = (bf16*)alloc(256 * 256 * 2);
  bf16* w3  = (bf16*)alloc(256 * 512 * 2);
  bf16* w1m = (bf16*)alloc(32 * 256 * 2);
  bf16* w2m = (bf16*)alloc(256 * 256 * 2);
  bf16* w3m = (bf16*)alloc(256 * 512 * 2);
  bf16* c1t = (bf16*)alloc(512 * 512 * 2);
  bf16* c1h = (bf16*)alloc(512 * 512 * 2);
  bf16* c1x = (bf16*)alloc(512 * 512 * 2);
  bf16* c2t = (bf16*)alloc(544 * 544 * 2);
  bf16* c2h = (bf16*)alloc(544 * 544 * 2);
  bf16* c2x = (bf16*)alloc(544 * 544 * 2);
  float* b2t = (float*)alloc(544 * 4);
  bf16* wi1 = (bf16*)alloc(512 * 512 * 2);
  bf16* wi2 = (bf16*)alloc(512 * 256 * 2);
  bf16* wi3 = (bf16*)alloc(256 * 256 * 2);
  bf16* wm1 = (bf16*)alloc(544 * 512 * 2);
  bf16* wm2 = (bf16*)alloc(512 * 256 * 2);
  bf16* wm3 = (bf16*)alloc(256 * 256 * 2);

  // activations / graph buffers
  bf16* enc   = (bf16*)alloc((size_t)P * 32 * 2);
  float* xdf  = (float*)alloc((size_t)P * 256 * 4);
  bf16* xdb   = (bf16*)alloc((size_t)P * 256 * 2);
  int* binidx = (int*)alloc((size_t)P * 4);
  int* order  = (int*)alloc((size_t)BATCH * NBINS * MAXPB * 4);
  int* counts = (int*)alloc((size_t)BATCH * NBINS * 4);
  float* dm   = (float*)alloc((size_t)BATCH * NBINS * MAXPB * MAXPB * 4);
  bf16* Ab    = (bf16*)alloc((size_t)BATCH * NBINS * MAXPB * MAXPB * 2);
  float* deg  = (float*)alloc((size_t)P * 4);
  float* nrm  = (float*)alloc((size_t)P * 4);
  bf16* bufA  = (bf16*)alloc((size_t)P * 544 * 2);
  bf16* bufB  = (bf16*)alloc((size_t)P * 544 * 2);
  bf16* bufC  = (bf16*)alloc((size_t)P * 544 * 2);
  bf16* tbf   = (bf16*)alloc((size_t)P * 544 * 2);
  float* spm  = (float*)alloc((size_t)P * 544 * 4);
  float* gate = (float*)alloc((size_t)P * 544 * 4);
  float* fhet = (float*)alloc((size_t)P * 544 * 4);
  float* oid  = (float*)alloc((size_t)P * 8 * 4);
  float* och  = (float*)alloc((size_t)P * 4);
  float* omom = (float*)alloc((size_t)P * 3 * 4);

  auto grid1 = [](int n) { return dim3((n + 255) / 256); };
  auto pack = [&](int wi, int K, int N, int Kp, int Np, bf16* dst) {
    pack_weight_t<<<grid1(Kp * Np), 256, 0, stream>>>(fin(wi), dst, K, N, Kp, Np);
  };
  // GEMM launcher: WT is [Np,Kpad]; ldo == Np
  auto gemm = [&](const bf16* A, int lda, const bf16* WT, const float* bias,
                  const float* rs, bf16* obf, float* of32, int Np, int Kpad,
                  int act) {
    dim3 g((P + 127) / 128, (Np + 63) / 64);
    gemm_bias_act<<<g, 256, 0, stream>>>(A, lda, WT, Kpad, bias, rs, obf, of32, Np,
                                         P, Np, Kpad, act);
  };

  // ---- pack weights -------------------------------------------------------
  pack(I_DIST_W, 26, 256, 32, 256, wd);
  pack(I1_W, 26, 256, 32, 256, w1);
  pack(I2_W, 256, 256, 256, 256, w2);
  pack(I3_W, 256, 512, 256, 512, w3);
  pack(I1M_W, 26, 256, 32, 256, w1m);
  pack(I2M_W, 256, 256, 256, 256, w2m);
  pack(I3M_W, 256, 512, 256, 512, w3m);
  pack(C1_WT, 512, 512, 512, 512, c1t);
  pack(C1_WH, 512, 512, 512, 512, c1h);
  pack(C1_TH, 512, 512, 512, 512, c1x);
  pack(C2_WT, 520, 520, 544, 544, c2t);
  pack(C2_WH, 520, 520, 544, 544, c2h);
  pack(C2_TH, 520, 520, 544, 544, c2x);
  pack_bias<<<grid1(544), 256, 0, stream>>>(fin(C2_BT), b2t, 520, 544);
  pack(ID1_W, 512, 512, 512, 512, wi1);
  pack(ID2_W, 512, 256, 512, 256, wi2);
  pack(ID3_W, 256, 256, 256, 256, wi3);
  pack(M1_W, 520, 512, 544, 512, wm1);
  pack(M2_W, 512, 256, 512, 256, wm2);
  pack(M3_W, 256, 256, 256, 256, wm3);

  // ---- encode + distance head --------------------------------------------
  encode_kernel<<<grid1(P * 32), 256, 0, stream>>>(X, enc, P);
  gemm(enc, 32, wd, fin(I_DIST_B), nullptr, xdb, xdf, 256, 32, 1);  // SELU

  // ---- LSH binning + adjacency -------------------------------------------
  bin_argmax<<<grid1(P), 256, 0, stream>>>(xdf, binidx, P);
  build_order<<<dim3(BATCH * NBINS), 32, 0, stream>>>(binidx, order, counts);
  gram_kernel<<<dim3(BATCH * NBINS, 16), 256, 0, stream>>>(xdb, order, dm);
  zero_f32<<<grid1(P), 256, 0, stream>>>(deg, P);
  adj_softmax<<<dim3(BATCH * NBINS), 256, 0, stream>>>(dm, Ab, order, counts, deg);
  norm_kernel<<<grid1(P), 256, 0, stream>>>(deg, nrm, P);

  // ---- classification branch ---------------------------------------------
  gemm(enc, 32, w1, fin(I1_B), nullptr, bufA, nullptr, 256, 32, 1);
  gemm(bufA, 256, w2, fin(I2_B), nullptr, bufB, nullptr, 256, 256, 1);
  gemm(bufB, 256, w3, fin(I3_B), nullptr, bufA, nullptr, 512, 256, 1);
  // GHConv #1 (H=512)
  gemm(bufA, 512, c1x, nullptr, nrm, tbf, nullptr, 512, 512, 0);  // (x@theta)*norm
  zero_f32<<<grid1(P * 512), 256, 0, stream>>>(spm, P * 512);
  spmm_kernel<<<dim3(BATCH * NBINS, 16), 256, 0, stream>>>(Ab, order, tbf, spm, 512);
  gemm(bufA, 512, c1h, nullptr, nullptr, nullptr, fhet, 512, 512, 0);
  gemm(bufA, 512, c1t, fin(C1_BT), nullptr, nullptr, gate, 512, 512, 2);
  ghconv_combine<<<grid1(P * 512), 256, 0, stream>>>(gate, fhet, spm, nrm, bufB, P,
                                                     512, 512);
  gemm(bufB, 512, wi1, fin(ID1_B), nullptr, bufA, nullptr, 512, 512, 1);
  gemm(bufA, 512, wi2, fin(ID2_B), nullptr, bufB, nullptr, 256, 512, 1);
  gemm(bufB, 256, wi3, fin(ID3_B), nullptr, bufA, nullptr, 256, 256, 1);
  small_dense<<<grid1(P * 8), 256, 0, stream>>>(bufA, 256, fin(OID_W), fin(OID_B),
                                                oid, P, 256, 8);
  small_dense<<<grid1(P), 256, 0, stream>>>(bufA, 256, fin(OCH_W), fin(OCH_B), och,
                                            P, 256, 1);

  // ---- momentum branch ----------------------------------------------------
  gemm(enc, 32, w1m, fin(I1M_B), nullptr, bufB, nullptr, 256, 32, 1);
  gemm(bufB, 256, w2m, fin(I2M_B), nullptr, bufC, nullptr, 256, 256, 1);
  gemm(bufC, 256, w3m, fin(I3M_B), nullptr, bufB, nullptr, 512, 256, 1);
  concat_kernel<<<grid1(P * 544), 256, 0, stream>>>(bufB, oid, bufC, P);
  // GHConv #2 (H=520, padded 544)
  gemm(bufC, 544, c2x, nullptr, nrm, tbf, nullptr, 544, 544, 0);
  zero_f32<<<grid1(P * 544), 256, 0, stream>>>(spm, P * 544);
  spmm_kernel<<<dim3(BATCH * NBINS, 16), 256, 0, stream>>>(Ab, order, tbf, spm, 544);
  gemm(bufC, 544, c2h, nullptr, nullptr, nullptr, fhet, 544, 544, 0);
  gemm(bufC, 544, c2t, b2t, nullptr, nullptr, gate, 544, 544, 2);
  ghconv_combine<<<grid1(P * 544), 256, 0, stream>>>(gate, fhet, spm, nrm, bufB, P,
                                                     520, 544);
  gemm(bufB, 544, wm1, fin(M1_B), nullptr, bufC, nullptr, 512, 544, 1);
  gemm(bufC, 512, wm2, fin(M2_B), nullptr, bufB, nullptr, 256, 512, 1);
  gemm(bufB, 256, wm3, fin(M3_B), nullptr, bufC, nullptr, 256, 256, 1);
  small_dense<<<grid1(P * 3), 256, 0, stream>>>(bufC, 256, fin(OM_W), fin(OM_B),
                                                omom, P, 256, 3);

  // ---- assemble output ----------------------------------------------------
  final_kernel<<<grid1(P * 12), 256, 0, stream>>>(X, oid, omom, och, (float*)d_out,
                                                  P);
}
